// Decoder_35244501631515
// MI455X (gfx1250) — compile-verified, matrix operand order checked by reading, not run on hardware
//
#include <hip/hip_runtime.h>
#include <hip/hip_bf16.h>

// ---------------------------------------------------------------------------
// GRU recurrence, persistent-workgroup formulation for MI455X (gfx1250).
//   B=256 batch rows are independent through the recurrence -> 16 workgroups
//   of 16 rows each run all 512 steps locally (no global sync at all).
//   Per-step GEMM tile: (16 x 512) @ (512 x 1536) via v_wmma_f32_16x16x32_f16.
//   K-chunk-outer loop: one A fragment (2x ds_load_b128) feeds 6 WMMAs.
// ---------------------------------------------------------------------------

typedef __attribute__((ext_vector_type(16))) _Float16 v16h;
typedef __attribute__((ext_vector_type(8)))  _Float16 v8h;
typedef __attribute__((ext_vector_type(8)))  float    v8f;

#define H       512                // hidden size (== H_IN)
#define NG      1536               // 3*H gate rows
#define ROWS    16                 // batch rows per workgroup (WMMA M)
#define NWG     16                 // 256 / ROWS
#define THREADS 512                // 16 wave32 per workgroup
#define KCHUNKS (H / 32)           // 16 K-chunks of 32 for f16 WMMA

__device__ __forceinline__ v8f wmma_f16(v16h a, v16h b, v8f c) {
    return __builtin_amdgcn_wmma_f32_16x16x32_f16(
        false, a, false, b, (short)0, c, false, false);
}

// 16 halves as two aligned 16B reads at half-offsets {0, +16}: matches the
// ISA 16-bit A/B fragment layout (lane halves hold complementary K subsets).
__device__ __forceinline__ v16h load_frag(const _Float16* p) {
    union { v16h v; v8h h[2]; } u;
    u.h[0] = *(const v8h*)(p);
    u.h[1] = *(const v8h*)(p + 16);
    return u.v;
}

__device__ __forceinline__ float sigmoid_f(float x) {
    return 1.0f / (1.0f + __expf(-x));
}
__device__ __forceinline__ float tanh_f(float x) {
    x = fminf(fmaxf(x, -15.0f), 15.0f);
    float e2 = __expf(2.0f * x);
    return (e2 - 1.0f) / (e2 + 1.0f);
}

// --- f32 -> f16 weight conversion (both W matrices then live in L2) ---------
__global__ void cvt_f32_f16(const float* __restrict__ src,
                            _Float16* __restrict__ dst, int n) {
    int i = blockIdx.x * blockDim.x + threadIdx.x;
    if (i < n) dst[i] = (_Float16)src[i];
}

// --- persistent GRU kernel --------------------------------------------------
__launch_bounds__(THREADS, 1)
__global__ void gru_persistent(const float* __restrict__ x,
                               const _Float16* __restrict__ Wih,  // (1536,512) f16
                               const _Float16* __restrict__ Whh,  // (1536,512) f16
                               const float* __restrict__ bih,
                               const float* __restrict__ bhh,
                               const int* __restrict__ seq_p,
                               float* __restrict__ out) {
    // double-buffered hidden-state tile, f16 row-major [ROWS][H]
    __shared__ __align__(16) _Float16 hbuf[2][ROWS * H];

    const int tid  = threadIdx.x;
    const int wave = tid >> 5;          // 0..15 (wave32)
    const int lane = tid & 31;
    const int half = lane >> 4;         // lane group 0 / 1
    const int m    = lane & 15;         // M row (A) / N col (B, C, D)
    const int b0   = blockIdx.x * ROWS;
    const int seq  = *seq_p;

    // ---- stage relu(x) tile as f16 into hbuf[0] (A source for gi GEMM) ----
    for (int i = tid; i < ROWS * H; i += THREADS) {
        int r = i >> 9;            // / H
        int c = i & (H - 1);
        float v = x[(b0 + r) * H + c];
        hbuf[0][i] = (_Float16)(v > 0.0f ? v : 0.0f);
    }
    __syncthreads();

    // wave w owns hidden columns j in [32w, 32w+32): j-tiles jtg0=2w, jtg1=2w+1.
    // Gate N-tiles jtg, 32+jtg, 64+jtg so r/z/n accumulators align elementwise.
    const int jtg0 = wave * 2;
    const int jtg1 = wave * 2 + 1;
    const int j0   = jtg0 * 16 + m;     // this lane's hidden column, tile 0
    const int j1   = jtg1 * 16 + m;     // this lane's hidden column, tile 1

    // Weight-row byte layouts identical for Wih / Whh: hoist offsets once.
    const size_t offR0 = ((size_t)(     jtg0) * 16 + m) * H + half * 8;
    const size_t offZ0 = ((size_t)(32 + jtg0) * 16 + m) * H + half * 8;
    const size_t offN0 = ((size_t)(64 + jtg0) * 16 + m) * H + half * 8;
    const size_t offR1 = ((size_t)(     jtg1) * 16 + m) * H + half * 8;
    const size_t offZ1 = ((size_t)(32 + jtg1) * 16 + m) * H + half * 8;
    const size_t offN1 = ((size_t)(64 + jtg1) * 16 + m) * H + half * 8;

    // ---- input-gate GEMM: gi = relu(x) @ W_ih.T  (kept in registers) ------
    v8f gR0 = {}, gZ0 = {}, gN0 = {}, gR1 = {}, gZ1 = {}, gN1 = {};
    {
        const _Float16* aB = &hbuf[0][m * H + half * 8];
#pragma unroll 2
        for (int kc = 0; kc < KCHUNKS; ++kc) {
            v16h A = load_frag(aB + kc * 32);
            gR0 = wmma_f16(A, load_frag(Wih + offR0 + kc * 32), gR0);
            gZ0 = wmma_f16(A, load_frag(Wih + offZ0 + kc * 32), gZ0);
            gN0 = wmma_f16(A, load_frag(Wih + offN0 + kc * 32), gN0);
            gR1 = wmma_f16(A, load_frag(Wih + offR1 + kc * 32), gR1);
            gZ1 = wmma_f16(A, load_frag(Wih + offZ1 + kc * 32), gZ1);
            gN1 = wmma_f16(A, load_frag(Wih + offN1 + kc * 32), gN1);
        }
    }
    // Fold biases: r,z gates absorb b_ih + b_hh; n gate absorbs b_ih only
    // (b_hh_n is multiplied by r each step, keep it separate).
    {
        const float br0 = bih[j0] + bhh[j0];
        const float bz0 = bih[H + j0] + bhh[H + j0];
        const float bn0 = bih[2 * H + j0];
        const float br1 = bih[j1] + bhh[j1];
        const float bz1 = bih[H + j1] + bhh[H + j1];
        const float bn1 = bih[2 * H + j1];
#pragma unroll
        for (int i = 0; i < 8; ++i) {
            gR0[i] += br0; gZ0[i] += bz0; gN0[i] += bn0;
            gR1[i] += br1; gZ1[i] += bz1; gN1[i] += bn1;
        }
    }
    const float bhn0 = bhh[2 * H + j0];
    const float bhn1 = bhh[2 * H + j1];
    float hp0[8], hp1[8];
#pragma unroll
    for (int i = 0; i < 8; ++i) { hp0[i] = 0.0f; hp1[i] = 0.0f; }

    __syncthreads();
    // ---- h0 = 0 ----
    for (int i = tid; i < ROWS * H; i += THREADS) hbuf[0][i] = (_Float16)0.0f;
    __syncthreads();

    // Hoist recurrent weight-row pointers (invariant across all steps).
    const _Float16* wR0 = Whh + offR0;
    const _Float16* wZ0 = Whh + offZ0;
    const _Float16* wN0 = Whh + offN0;
    const _Float16* wR1 = Whh + offR1;
    const _Float16* wZ1 = Whh + offZ1;
    const _Float16* wN1 = Whh + offN1;

    const size_t bstride = (size_t)seq * H;   // out is (B, seq, H)

    // ---- seq serially dependent steps, one barrier each -------------------
    int p = 0;
    for (int t = 0; t < seq; ++t) {
        const _Float16* aB   = &hbuf[p][m * H + half * 8];
        _Float16*       hNew = hbuf[1 - p];

        v8f aR0 = {}, aZ0 = {}, aN0 = {}, aR1 = {}, aZ1 = {}, aN1 = {};
#pragma unroll 2
        for (int kc = 0; kc < KCHUNKS; ++kc) {
            v16h A = load_frag(aB + kc * 32);       // ds_load_b128 x2, feeds 6 WMMA
            aR0 = wmma_f16(A, load_frag(wR0 + kc * 32), aR0);
            aZ0 = wmma_f16(A, load_frag(wZ0 + kc * 32), aZ0);
            aN0 = wmma_f16(A, load_frag(wN0 + kc * 32), aN0);
            aR1 = wmma_f16(A, load_frag(wR1 + kc * 32), aR1);
            aZ1 = wmma_f16(A, load_frag(wZ1 + kc * 32), aZ1);
            aN1 = wmma_f16(A, load_frag(wN1 + kc * 32), aN1);
        }

        // GRU nonlinearity, register math (C/D layout: lane=N col, vgpr=M row)
#pragma unroll
        for (int i = 0; i < 8; ++i) {
            const int b = i + half * 8;
            float rg = sigmoid_f(gR0[i] + aR0[i]);
            float zg = sigmoid_f(gZ0[i] + aZ0[i]);
            float ng = tanh_f(gN0[i] + rg * (aN0[i] + bhn0));
            float hn = (1.0f - zg) * ng + zg * hp0[i];
            hp0[i] = hn;
            hNew[b * H + j0] = (_Float16)hn;
            out[(size_t)(b0 + b) * bstride + (size_t)t * H + j0] = hn;
        }
#pragma unroll
        for (int i = 0; i < 8; ++i) {
            const int b = i + half * 8;
            float rg = sigmoid_f(gR1[i] + aR1[i]);
            float zg = sigmoid_f(gZ1[i] + aZ1[i]);
            float ng = tanh_f(gN1[i] + rg * (aN1[i] + bhn1));
            float hn = (1.0f - zg) * ng + zg * hp1[i];
            hp1[i] = hn;
            hNew[b * H + j1] = (_Float16)hn;
            out[(size_t)(b0 + b) * bstride + (size_t)t * H + j1] = hn;
        }

        __syncthreads();   // hNew complete before it becomes the A source
        p ^= 1;
    }
}

extern "C" void kernel_launch(void* const* d_in, const int* in_sizes, int n_in,
                              void* d_out, int out_size, void* d_ws, size_t ws_size,
                              hipStream_t stream) {
    const float* x    = (const float*)d_in[0];
    const float* Wih  = (const float*)d_in[1];
    const float* Whh  = (const float*)d_in[2];
    const float* bih  = (const float*)d_in[3];
    const float* bhh  = (const float*)d_in[4];
    const int*   seqp = (const int*)d_in[5];

    _Float16* Wih16 = (_Float16*)d_ws;
    _Float16* Whh16 = Wih16 + (size_t)NG * H;

    const int nW = NG * H;
    cvt_f32_f16<<<(nW + 255) / 256, 256, 0, stream>>>(Wih, Wih16, nW);
    cvt_f32_f16<<<(nW + 255) / 256, 256, 0, stream>>>(Whh, Whh16, nW);

    gru_persistent<<<NWG, THREADS, 0, stream>>>(x, Wih16, Whh16, bih, bhh,
                                                seqp, (float*)d_out);
}
